// Net_52183852646957
// MI455X (gfx1250) — matrix operand order repaired and checked by reading, not compiled, not run on hardware
//
#include <hip/hip_runtime.h>
#include <hip/hip_fp16.h>

// Fixed problem dims from the reference
#define S_N   5
#define BATCH 2048
#define KDIM  128
#define DDIM  64
#define H1D   256
#define H2D   128

typedef __attribute__((ext_vector_type(16))) _Float16 v16h;
typedef __attribute__((ext_vector_type(8)))  float    v8f;
typedef __attribute__((ext_vector_type(4)))  float    f32x4;

union HFrag { v16h v; _Float16 h[16]; };

__device__ __forceinline__ float fast_rcp(float x) {
#if __has_builtin(__builtin_amdgcn_rcpf)
  return __builtin_amdgcn_rcpf(x);     // v_rcp_f32
#else
  return 1.f / x;
#endif
}
__device__ __forceinline__ float fast_tanh(float x) {
#if __has_builtin(__builtin_amdgcn_tanhf)
  return __builtin_amdgcn_tanhf(x);    // gfx1250 v_tanh_f32 (TRANS32)
#else
  // branch-free: tanh(x) = 1 - 2/(exp(2x)+1); saturates correctly at +/-inf
  float e = __expf(2.f * x);
  return 1.f - 2.f * fast_rcp(e + 1.f);
#endif
}
__device__ __forceinline__ float softplus_f(float x) {
  return (x > 20.f) ? x : log1pf(__expf(x));
}
__device__ __forceinline__ float sigmoid_f(float x) {
  return fast_rcp(1.f + __expf(-x));
}

// ---------------------------------------------------------------------------
// Kernel 0: parameter-only work, done once per launch.
//   * std tables: 1e-6 + softplus(eta)  (so the 2048 batch blocks don't each
//     recompute 32K softplus -> halves transcendental load of the hot kernel)
//   * KL( N(mu,std) || N(0,1) ) summed over all weight/bias params -> d_out tail
// ---------------------------------------------------------------------------
__global__ void kl_std_kernel(
    const float* w_mu1, const float* w_eta1, const float* b_mu1, const float* b_eta1,
    const float* w_mu2, const float* w_eta2, const float* b_mu2, const float* b_eta2,
    const float* w_mu3, const float* w_eta3, const float* b_mu3, const float* b_eta3,
    float* std_w1, float* std_b1, float* std_w2, float* std_b2,
    float* std_w3, float* std_b3, float* kl_out) {
  const int t = threadIdx.x;
  const int NT = 256;
  float acc = 0.f;
  auto seg = [&](const float* mu, const float* eta, int n, float* so) {
    for (int i = t; i < n; i += NT) {
      float s = 1e-6f + softplus_f(eta[i]);
      so[i] = s;
      float m = mu[i];
      acc += -logf(s) + 0.5f * (s * s + m * m - 1.f);
    }
  };
  seg(w_mu1, w_eta1, H1D * KDIM, std_w1);
  seg(b_mu1, b_eta1, H1D,        std_b1);
  seg(w_mu2, w_eta2, H2D * H1D,  std_w2);
  seg(b_mu2, b_eta2, H2D,        std_b2);
  seg(w_mu3, w_eta3, H2D,        std_w3);
  seg(b_mu3, b_eta3, 1,          std_b3);

  __shared__ float red[256];
  red[t] = acc;
  __syncthreads();
  for (int off = 128; off > 0; off >>= 1) {
    if (t < off) red[t] += red[t + off];
    __syncthreads();
  }
  if (t == 0) kl_out[0] = red[0];
}

// ---------------------------------------------------------------------------
// Fused main kernel: one block (256 threads = 8 wave32) per batch element b.
// Phase 1-2: gathers + 4 small dot products over D=64 -> sampled x -> LDS
//            tile xT[16 x 128] f16 (rows 5..15 zero padded for WMMA M=16).
// Phase 3:   layer1: 16 N-tiles of H1=256, 2 per wave; B fragments built on
//            the fly: w = exp(mu + std*eps)  (streams eps_w1: 256 MiB total).
//            v_wmma_f32_16x16x32_f16 accumulate, K=128 in 4 steps.
// Phase 4:   layer2: 8 N-tiles of H2=128, 1 per wave, K=256 in 8 steps.
// Phase 5:   layer3 (O=1): plain dot + sigmoid, write out[s*B + b].
// ---------------------------------------------------------------------------
__global__ __launch_bounds__(256)
void fused_net_kernel(
    const int* stu_id, const int* exer_id, const float* kn_r,
    const float* stu_cnt, const float* exer_cnt,
    const float* stu_emb, const float* exer_emb, const float* kn_emb,
    const float* e_disc_mean, const float* e_disc_eta,
    const float* stat_mean_w, const float* stat_mean_b,
    const float* stat_std_w, const float* stat_std_b,
    const float* kdiff_mean_w, const float* kdiff_mean_b,
    const float* kdiff_std_w, const float* kdiff_std_b,
    const float* lam1_stu, const float* lam2_stu,
    const float* lam1_exer, const float* lam2_exer,
    const float* w_mu1, const float* b_mu1,
    const float* w_mu2, const float* b_mu2,
    const float* w_mu3, const float* b_mu3,
    const float* std_w1, const float* std_b1,
    const float* std_w2, const float* std_b2,
    const float* std_w3, const float* std_b3,
    const float* eps_stat, const float* eps_kdiff, const float* eps_disc,
    const float* eps_w1, const float* eps_b1,
    const float* eps_w2, const float* eps_b2,
    const float* eps_w3, const float* eps_b3,
    float* out) {
  __shared__ __align__(16) _Float16 xT[16 * KDIM];   // layer1 A tile
  __shared__ __align__(16) _Float16 h1T[16 * H1D];   // layer2 A tile
  __shared__ __align__(16) _Float16 h2T[16 * H2D];   // layer3 input
  __shared__ float suS[DDIM], eeS[DDIM];
  __shared__ float smwS[DDIM], sswS[DDIM], kdwS[DDIM], kswS[DDIM];
  __shared__ float pbuf[S_N * H2D];

  const int b = blockIdx.x;
  const int t = threadIdx.x;
  const int si = stu_id[b], ei = exer_id[b];

  // ---- Phase 1: stage embeddings + projection vectors in LDS
  if (t < DDIM) {
    suS[t]  = stu_emb[(size_t)si * DDIM + t];
    smwS[t] = stat_mean_w[t];
    sswS[t] = stat_std_w[t];
  } else if (t < 2 * DDIM) {
    int d = t - DDIM;
    eeS[d]  = exer_emb[(size_t)ei * DDIM + d];
    kdwS[d] = kdiff_mean_w[d];
    kswS[d] = kdiff_std_w[d];
  }
  __syncthreads();

  // ---- Phase 2: per-k stats -> sampled x tile (f16, rows >= S_N zeroed)
  if (t < KDIM) {
    const int k = t;
    const float* km = kn_emb + (size_t)k * DDIM;
    float sm = 0.f, ss = 0.f, dm = 0.f, ds = 0.f;
    for (int d = 0; d < DDIM; ++d) {
      float kv = km[d];
      float pu = suS[d] * kv, pe = eeS[d] * kv;
      sm += pu * smwS[d];  ss += pu * sswS[d];
      dm += pe * kdwS[d];  ds += pe * kswS[d];
    }
    sm += stat_mean_b[0];
    dm += kdiff_mean_b[0];
    float ssd = softplus_f(lam1_stu[0])  * __expf(-softplus_f(lam2_stu[0])  * stu_cnt[si]);
    float dsd = softplus_f(lam1_exer[0]) * __expf(-softplus_f(lam2_exer[0]) * exer_cnt[ei]);
    float sstd = ssd + softplus_f(ss + stat_std_b[0]);
    float dstd = dsd + softplus_f(ds + kdiff_std_b[0]);
    float dmean = e_disc_mean[ei];
    float dsig  = softplus_f(e_disc_eta[ei]);
    float knr   = kn_r[(size_t)b * KDIM + k];
    #pragma unroll
    for (int s = 0; s < S_N; ++s) {
      float es = eps_stat [((size_t)s * BATCH + b) * KDIM + k];
      float ek = eps_kdiff[((size_t)s * BATCH + b) * KDIM + k];
      float ed = eps_disc [(size_t)s * BATCH + b];
      float stat = sigmoid_f(sm + sstd * es);
      float kdf  = sigmoid_f(dm + dstd * ek);
      float disc = sigmoid_f(dmean + dsig * ed);
      xT[s * KDIM + k] = (_Float16)(disc * (stat - kdf) * knr);
    }
    for (int m = S_N; m < 16; ++m) xT[m * KDIM + k] = (_Float16)0.f;
  }
  __syncthreads();

  const int wid  = t >> 5;
  const int lane = t & 31;
  const int nn   = lane & 15;   // N column (also A row M for A fragments)
  const int kh   = lane >> 4;   // lane half -> K sub-block select

  // ---- Phase 3: layer1 (16x256 out, K=128), 2 N-tiles per wave
  #pragma unroll
  for (int nt = wid * 2; nt <= wid * 2 + 1; ++nt) {
    const int o = nt * 16 + nn;
    const float* er = eps_w1 + ((size_t)b * H1D + o) * KDIM;
    const float* mr = w_mu1  + (size_t)o * KDIM;
    const float* sr = std_w1 + (size_t)o * KDIM;
    v8f acc = {};
    #pragma unroll
    for (int ks = 0; ks < KDIM; ks += 32) {
      HFrag a;  // A layout: lanes<16 K=[0..15]; lanes>=16 K=[16..31]
      __builtin_memcpy(&a.h[0], &xT[nn * KDIM + ks + 16 * kh], 16);
      __builtin_memcpy(&a.h[8], &xT[nn * KDIM + ks + 16 * kh + 8], 16);
      const int kk = ks + 16 * kh;  // B uses A's contiguous k order: lane-half picks K 16-group
      const f32x4* e4 = (const f32x4*)(er + kk);
      const f32x4* m4 = (const f32x4*)(mr + kk);
      const f32x4* s4 = (const f32x4*)(sr + kk);
      HFrag bf;
      #pragma unroll
      for (int q = 0; q < 4; ++q) {
        f32x4 ev = e4[q], mv = m4[q], sv = s4[q];
        #pragma unroll
        for (int j = 0; j < 4; ++j)
          bf.h[4 * q + j] = (_Float16)__expf(mv[j] + sv[j] * ev[j]);
      }
      acc = __builtin_amdgcn_wmma_f32_16x16x32_f16(
          false, a.v, false, bf.v, (short)0, acc, false, false);
    }
    const float bias = b_mu1[o] + std_b1[o] * eps_b1[(size_t)b * H1D + o];
    #pragma unroll
    for (int r = 0; r < 8; ++r) {
      const int m = r + 8 * kh;  // C/D layout: vgpr r, lane-half picks M+8
      float v = (m < S_N) ? fast_tanh(acc[r] + bias) : 0.f;
      h1T[m * H1D + o] = (_Float16)v;
    }
  }
  __syncthreads();

  // ---- Phase 4: layer2 (16x128 out, K=256), 1 N-tile per wave
  {
    const int o = wid * 16 + nn;
    const float* er = eps_w2 + ((size_t)b * H2D + o) * H1D;
    const float* mr = w_mu2  + (size_t)o * H1D;
    const float* sr = std_w2 + (size_t)o * H1D;
    v8f acc = {};
    #pragma unroll
    for (int ks = 0; ks < H1D; ks += 32) {
      HFrag a;
      __builtin_memcpy(&a.h[0], &h1T[nn * H1D + ks + 16 * kh], 16);
      __builtin_memcpy(&a.h[8], &h1T[nn * H1D + ks + 16 * kh + 8], 16);
      const int kk = ks + 16 * kh;
      const f32x4* e4 = (const f32x4*)(er + kk);
      const f32x4* m4 = (const f32x4*)(mr + kk);
      const f32x4* s4 = (const f32x4*)(sr + kk);
      HFrag bf;
      #pragma unroll
      for (int q = 0; q < 4; ++q) {
        f32x4 ev = e4[q], mv = m4[q], sv = s4[q];
        #pragma unroll
        for (int j = 0; j < 4; ++j)
          bf.h[4 * q + j] = (_Float16)__expf(mv[j] + sv[j] * ev[j]);
      }
      acc = __builtin_amdgcn_wmma_f32_16x16x32_f16(
          false, a.v, false, bf.v, (short)0, acc, false, false);
    }
    const float bias = b_mu2[o] + std_b2[o] * eps_b2[(size_t)b * H2D + o];
    #pragma unroll
    for (int r = 0; r < 8; ++r) {
      const int m = r + 8 * kh;
      float v = (m < S_N) ? fast_tanh(acc[r] + bias) : 0.f;
      h2T[m * H2D + o] = (_Float16)v;
    }
  }
  __syncthreads();

  // ---- Phase 5: layer3 (O=1): sampled-weight dot + sigmoid
  if (t < H2D) {
    float w3 = __expf(w_mu3[t] + std_w3[t] * eps_w3[(size_t)b * H2D + t]);
    #pragma unroll
    for (int s = 0; s < S_N; ++s)
      pbuf[s * H2D + t] = (float)h2T[s * H2D + t] * w3;
  }
  __syncthreads();
  if (t < S_N) {
    float sum = 0.f;
    for (int i = 0; i < H2D; ++i) sum += pbuf[t * H2D + i];
    float bias = b_mu3[0] + std_b3[0] * eps_b3[b];
    out[(size_t)t * BATCH + b] = sigmoid_f(sum + bias);
  }
}

// ---------------------------------------------------------------------------
extern "C" void kernel_launch(void* const* d_in, const int* in_sizes, int n_in,
                              void* d_out, int out_size, void* d_ws, size_t ws_size,
                              hipStream_t stream) {
  (void)in_sizes; (void)n_in; (void)out_size; (void)ws_size;
  #define FIN(i) ((const float*)d_in[(i)])
  const int* stu_id  = (const int*)d_in[0];
  const int* exer_id = (const int*)d_in[1];

  // workspace layout (floats): softplus std tables
  float* ws     = (float*)d_ws;
  float* std_w1 = ws;                 // H1*K   = 32768
  float* std_b1 = ws + 32768;         // H1     = 256
  float* std_w2 = ws + 33024;         // H2*H1  = 32768
  float* std_b2 = ws + 65792;         // H2     = 128
  float* std_w3 = ws + 65920;         // H2     = 128
  float* std_b3 = ws + 66048;         // 1
  float* out    = (float*)d_out;      // [S*B] probs then [1] KL

  kl_std_kernel<<<1, 256, 0, stream>>>(
      FIN(22), FIN(23), FIN(24), FIN(25),
      FIN(26), FIN(27), FIN(28), FIN(29),
      FIN(30), FIN(31), FIN(32), FIN(33),
      std_w1, std_b1, std_w2, std_b2, std_w3, std_b3,
      out + (size_t)S_N * BATCH);

  fused_net_kernel<<<BATCH, 256, 0, stream>>>(
      stu_id, exer_id, FIN(2), FIN(3), FIN(4),
      FIN(5), FIN(6), FIN(7), FIN(8), FIN(9),
      FIN(10), FIN(11), FIN(12), FIN(13),
      FIN(14), FIN(15), FIN(16), FIN(17),
      FIN(18), FIN(19), FIN(20), FIN(21),
      FIN(22), FIN(24), FIN(26), FIN(28), FIN(30), FIN(32),
      std_w1, std_b1, std_w2, std_b2, std_w3, std_b3,
      FIN(34), FIN(35), FIN(36),
      FIN(37), FIN(38), FIN(39), FIN(40), FIN(41), FIN(42),
      out);
  #undef FIN
}
